// BranchPredictionGNN_12326556139937
// MI455X (gfx1250) — matrix operand order repaired and checked by reading, not compiled, not run on hardware
//
#include <hip/hip_runtime.h>
#include <hip/hip_bf16.h>
#include <stdint.h>

// ---------------------------------------------------------------------------
// CDNA5 (gfx1250) GNN pipeline: bf16 WMMA 16x16x32 for all dense GEMMs,
// float atomics for segment sums, fused edge-head kernel.
// Round 2: use native f32->bf16 conversion hardware (v_cvt) instead of
// software round-to-nearest-even bit tricks, shrinking the VALU chain that
// feeds each WMMA A-fragment.
// ---------------------------------------------------------------------------

typedef __attribute__((ext_vector_type(16))) __bf16 v16bf;
typedef __attribute__((ext_vector_type(8)))  float  v8f;

#define NODE_DIM 128
#define H 64

// Native conversion: compiler lowers to v_cvt(_pk)_bf16_f32 on gfx1250.
__device__ __forceinline__ unsigned pack2(float a, float b) {
  __bf16 lo = (__bf16)a, hi = (__bf16)b;
  unsigned short ul = __builtin_bit_cast(unsigned short, lo);
  unsigned short uh = __builtin_bit_cast(unsigned short, hi);
  return (unsigned)ul | ((unsigned)uh << 16);
}

union FragU { unsigned u[8]; uint4 q[2]; v16bf v; };

// A-matrix 16x32 bf16 fragment (this code's contiguous k order, matching pack_b, not the ISA table order): lane L (and L+16) hold row M=L&15;
// lanes 0-15: K = {0..15}; lanes 16-31: K = {16..31};
// vector element e<8 -> K = o0+e ; element 8+e -> K = o1+e
// (pairs (2j,2j+1) land in VGPR j low/high halves).
__device__ __forceinline__ v16bf make_afrag(const float* row, int kbase, int half, int kmax) {
  v16bf v;
  int o0 = kbase + half * 16;
  int o1 = kbase + half * 16 + 8;
#pragma unroll
  for (int j = 0; j < 8; ++j) {
    int k0 = o0 + j;
    v[j] = (__bf16)((k0 < kmax) ? row[k0] : 0.f);
    int k1 = o1 + j;
    v[8 + j] = (__bf16)((k1 < kmax) ? row[k1] : 0.f);
  }
  return v;
}

__device__ __forceinline__ v16bf load_bfrag(const unsigned* p) {
  FragU U;
  const uint4* q = (const uint4*)p;
  U.q[0] = q[0];
  U.q[1] = q[1];
  return U.v;
}

__device__ __forceinline__ v8f wmma_bf16(v16bf a, v16bf b, v8f c) {
  return __builtin_amdgcn_wmma_f32_16x16x32_bf16(false, a, false, b, (short)0, c, false, false);
}

// ---------------------------------------------------------------------------
// Weight pre-pack into B-fragment layout (B 32x16: lanes 0-15 K=0..15,
// lanes 16-31 K=16..31, VGPR j packs K = half*16 + 2j, 2j+1 for column N=lane&15).
// Logical weight Wcat[k][n] = (n < split) ? A[k*lda+n] : B[k*ldb+n-split], zero for k>=K.
// Frag index = t*Kt + kt ; 256 dwords per fragment (32 lanes x 8 dwords).
// ---------------------------------------------------------------------------
__global__ void pack_b(const float* __restrict__ A, const float* __restrict__ B,
                       int lda, int ldb, int split, int K, int Kt, int Nt,
                       unsigned* __restrict__ dst) {
  int tid = blockIdx.x * blockDim.x + threadIdx.x;
  if (tid >= Nt * Kt * 32) return;
  int lane = tid & 31;
  int frag = tid >> 5;
  int kt = frag % Kt, t = frag / Kt;
  int n  = t * 16 + (lane & 15);
  int kb = kt * 32 + (lane >> 4) * 16;
  unsigned* d = dst + frag * 256 + lane * 8;
#pragma unroll
  for (int j = 0; j < 8; ++j) {
    int k0 = kb + 2 * j, k1 = k0 + 1;
    float v0 = 0.f, v1 = 0.f;
    if (k0 < K) v0 = (n < split) ? A[(long)k0 * lda + n] : B[(long)k0 * ldb + (n - split)];
    if (k1 < K) v1 = (n < split) ? A[(long)k1 * lda + n] : B[(long)k1 * ldb + (n - split)];
    d[j] = pack2(v0, v1);
  }
}

// ---------------------------------------------------------------------------
// Utility kernels
// ---------------------------------------------------------------------------
__global__ void zero_f(float* __restrict__ p, long n) {
  long i = (long)blockIdx.x * blockDim.x + threadIdx.x;
  if (i < n) p[i] = 0.f;
}

__global__ void deg_count(const int* __restrict__ tgt, float* __restrict__ cnt, int E) {
  int e = blockIdx.x * blockDim.x + threadIdx.x;
  if (e < E) atomicAdd(cnt + tgt[e], 1.0f);
}

// acc[tgt[e]][f] += feat[src[e]][f]   (64 floats per edge)
__global__ void scatter_add64(const float* __restrict__ feat, const int* __restrict__ src,
                              const int* __restrict__ tgt, float* __restrict__ acc, int E) {
  long gid = (long)blockIdx.x * blockDim.x + threadIdx.x;
  if (gid >= (long)E * H) return;
  int e = (int)(gid >> 6), f = (int)(gid & 63);
  atomicAdd(acc + (long)tgt[e] * H + f, feat[(long)src[e] * H + f]);
}

// h = relu(s / max(cnt,1) + bias + z)
__global__ void sage_finish(const float* __restrict__ s, const float* __restrict__ cnt,
                            const float* __restrict__ bias, const float* __restrict__ z,
                            float* __restrict__ h, int N) {
  long gid = (long)blockIdx.x * blockDim.x + threadIdx.x;
  if (gid >= (long)N * H) return;
  int n = (int)(gid >> 6), f = (int)(gid & 63);
  float c = cnt[n];
  c = c > 1.f ? c : 1.f;
  h[gid] = fmaxf(s[gid] / c + bias[f] + z[gid], 0.f);
}

// ---------------------------------------------------------------------------
// Node GEMM: out0 = in @ Wcols[0:64], out1 = in @ Wcols[64:128]
// One wave per 16-row tile; KT in {2,4} (K = 64 or 128); Nt fixed = 8.
// ---------------------------------------------------------------------------
template <int KT>
__global__ __launch_bounds__(256) void node_gemm(const float* __restrict__ in, int K,
                                                 const unsigned* __restrict__ frags,
                                                 float* __restrict__ out0,
                                                 float* __restrict__ out1, int nRows) {
  int wave = threadIdx.x >> 5;
  int lane = threadIdx.x & 31;
  int tile = blockIdx.x * 8 + wave;
  int nTiles = (nRows + 15) >> 4;
  if (tile >= nTiles) return;                 // wave-uniform
  int l15 = lane & 15, half = lane >> 4;
  int row = tile * 16 + l15;
  if (row >= nRows) row = nRows - 1;
  const float* rp = in + (long)row * K;

  v16bf a[KT];
#pragma unroll
  for (int kt = 0; kt < KT; ++kt) a[kt] = make_afrag(rp, kt * 32, half, K);

#pragma unroll
  for (int t = 0; t < 8; ++t) {
    v8f c = {};
#pragma unroll
    for (int kt = 0; kt < KT; ++kt) {
      v16bf b = load_bfrag(frags + (((t * KT + kt) * 32) + lane) * 8);
      c = wmma_bf16(a[kt], b, c);
    }
    float* op = (t < 4) ? out0 : out1;
    int col = (t & 3) * 16 + l15;
    int rb  = tile * 16 + half * 8;
#pragma unroll
    for (int r = 0; r < 8; ++r) {
      int rr = rb + r;
      if (rr < nRows) op[(long)rr * H + col] = c[r];
    }
  }
}

// ---------------------------------------------------------------------------
// Fused edge head: ea = (relu(eattr@We1+be1))@We2+be2 ; ef = h1[src]+h1[tgt]+ea ;
// out = sigmoid(relu(ef@Wo1+bo1)@Wo2+bo2).  One wave per 16-edge tile.
// LDS re-staging converts WMMA C-layout -> A-layout between stages.
// ---------------------------------------------------------------------------
__global__ __launch_bounds__(256) void edge_head(
    const float* __restrict__ eattr, const int* __restrict__ src, const int* __restrict__ tgt,
    const float* __restrict__ h1,
    const unsigned* __restrict__ fWe1, const float* __restrict__ be1,
    const unsigned* __restrict__ fWe2, const float* __restrict__ be2,
    const unsigned* __restrict__ fWo1, const float* __restrict__ bo1,
    const float* __restrict__ Wo2, const float* __restrict__ bo2,
    float* __restrict__ out, int E) {
  __shared__ float sbuf[8][16 * H];
  __shared__ int   sidx[8][32];
  int wave = threadIdx.x >> 5, lane = threadIdx.x & 31;
  int tile = blockIdx.x * 8 + wave;
  int nTiles = (E + 15) >> 4;
  if (tile >= nTiles) return;                 // wave-uniform
  float* sb = sbuf[wave];
  int*   si = sidx[wave];
  int l15 = lane & 15, half = lane >> 4;
  int ebase = tile * 16;

  if (lane < 16) {
    int e = ebase + lane;
    if (e >= E) e = E - 1;
    si[lane]      = src[e];
    si[16 + lane] = tgt[e];
  }
  __builtin_amdgcn_wave_barrier();

  // 1) t1 = relu(edge_attr @ We1 + be1)   (K=16 zero-padded to 32)
  {
    int e = ebase + l15;
    if (e >= E) e = E - 1;
    const float* er = eattr + (long)e * 16;
    v16bf aE = make_afrag(er, 0, half, 16);
#pragma unroll
    for (int t = 0; t < 4; ++t) {
      v8f c = {};
      c = wmma_bf16(aE, load_bfrag(fWe1 + (t * 32 + lane) * 8), c);
      float bias = be1[t * 16 + l15];
#pragma unroll
      for (int r = 0; r < 8; ++r)
        sb[(half * 8 + r) * H + t * 16 + l15] = fmaxf(c[r] + bias, 0.f);
    }
  }
  __builtin_amdgcn_wave_barrier();

  // 2) ea = t1 @ We2 + be2  (K=64, kept in registers)
  v8f ea[4];
  {
    v16bf a0 = make_afrag(sb + l15 * H, 0, half, 64);
    v16bf a1 = make_afrag(sb + l15 * H, 32, half, 64);
#pragma unroll
    for (int t = 0; t < 4; ++t) {
      v8f c = {};
      c = wmma_bf16(a0, load_bfrag(fWe2 + ((t * 2 + 0) * 32 + lane) * 8), c);
      c = wmma_bf16(a1, load_bfrag(fWe2 + ((t * 2 + 1) * 32 + lane) * 8), c);
      float bias = be2[t * 16 + l15];
#pragma unroll
      for (int r = 0; r < 8; ++r) ea[t][r] = c[r] + bias;
    }
  }
  __builtin_amdgcn_wave_barrier();

  // 3) ef = ea + h1[src] + h1[tgt]  (stage in LDS, cooperative gather)
#pragma unroll
  for (int t = 0; t < 4; ++t)
#pragma unroll
    for (int r = 0; r < 8; ++r)
      sb[(half * 8 + r) * H + t * 16 + l15] = ea[t][r];
  __builtin_amdgcn_wave_barrier();
  for (int f = lane; f < 16 * H; f += 32) {
    int e = f >> 6, c = f & 63;
    sb[f] += h1[(long)si[e] * H + c] + h1[(long)si[16 + e] * H + c];
  }
  __builtin_amdgcn_wave_barrier();

  // 4) g = relu(ef @ Wo1 + bo1)
  v8f g[4];
  {
    v16bf a0 = make_afrag(sb + l15 * H, 0, half, 64);
    v16bf a1 = make_afrag(sb + l15 * H, 32, half, 64);
#pragma unroll
    for (int t = 0; t < 4; ++t) {
      v8f c = {};
      c = wmma_bf16(a0, load_bfrag(fWo1 + ((t * 2 + 0) * 32 + lane) * 8), c);
      c = wmma_bf16(a1, load_bfrag(fWo1 + ((t * 2 + 1) * 32 + lane) * 8), c);
      float bias = bo1[t * 16 + l15];
#pragma unroll
      for (int r = 0; r < 8; ++r) g[t][r] = fmaxf(c[r] + bias, 0.f);
    }
  }
  __builtin_amdgcn_wave_barrier();
#pragma unroll
  for (int t = 0; t < 4; ++t)
#pragma unroll
    for (int r = 0; r < 8; ++r)
      sb[(half * 8 + r) * H + t * 16 + l15] = g[t][r];
  __builtin_amdgcn_wave_barrier();

  // 5) out = sigmoid(g . Wo2 + bo2)
  if (lane < 16) {
    int e = ebase + lane;
    if (e < E) {
      float acc = bo2[0];
#pragma unroll 8
      for (int n = 0; n < H; ++n) acc += sb[lane * H + n] * Wo2[n];
      out[e] = 1.f / (1.f + __expf(-acc));
    }
  }
}

// ---------------------------------------------------------------------------
// Host-side orchestration
// ---------------------------------------------------------------------------
extern "C" void kernel_launch(void* const* d_in, const int* in_sizes, int n_in,
                              void* d_out, int out_size, void* d_ws, size_t ws_size,
                              hipStream_t stream) {
  (void)n_in; (void)out_size; (void)ws_size;
  const float* x   = (const float*)d_in[0];
  const int*   ei  = (const int*)d_in[1];
  const float* ea  = (const float*)d_in[2];
  const float* We1 = (const float*)d_in[3];
  const float* be1 = (const float*)d_in[4];
  const float* We2 = (const float*)d_in[5];
  const float* be2 = (const float*)d_in[6];
  const float* Wl0 = (const float*)d_in[7];
  const float* bl0 = (const float*)d_in[8];
  const float* Wr0 = (const float*)d_in[9];
  const float* Wl1 = (const float*)d_in[10];
  const float* bl1 = (const float*)d_in[11];
  const float* Wr1 = (const float*)d_in[12];
  const float* Wo1 = (const float*)d_in[13];
  const float* bo1 = (const float*)d_in[14];
  const float* Wo2 = (const float*)d_in[15];
  const float* bo2 = (const float*)d_in[16];
  float* out = (float*)d_out;

  int N = in_sizes[0] / NODE_DIM;
  int E = in_sizes[1] / 2;
  const int* src = ei;
  const int* tgt = ei + E;

  // Workspace layout (floats): cnt[N] | S[N*64] | Y[N*64] | Z[N*64] | W[N*64] | frags
  float* ws   = (float*)d_ws;
  float* cnt  = ws;
  float* bufS = cnt + N;
  float* bufY = bufS + (long)N * H;
  float* bufZ = bufY + (long)N * H;
  float* bufW = bufZ + (long)N * H;
  unsigned* frg  = (unsigned*)(bufW + (long)N * H);
  unsigned* fW0  = frg;               // [Wl0|Wr0] 128x128 : 8*4*256 dwords
  unsigned* fW1  = fW0 + 8 * 4 * 256; // [Wl1|Wr1]  64x128 : 8*2*256
  unsigned* fWe1 = fW1 + 8 * 2 * 256; // We1 16x64 (K pad)  : 4*1*256
  unsigned* fWe2 = fWe1 + 4 * 1 * 256;// We2 64x64          : 4*2*256
  unsigned* fWo1 = fWe2 + 4 * 2 * 256;// Wo1 64x64          : 4*2*256

  dim3 blk(256);
  long zc = (long)N * H + N;  // cnt + bufS are contiguous
  zero_f<<<(unsigned)((zc + 255) / 256), blk, 0, stream>>>(ws, zc);

  pack_b<<<(8 * 4 * 32 + 255) / 256, blk, 0, stream>>>(Wl0, Wr0, 64, 64, 64, 128, 4, 8, fW0);
  pack_b<<<(8 * 2 * 32 + 255) / 256, blk, 0, stream>>>(Wl1, Wr1, 64, 64, 64,  64, 2, 8, fW1);
  pack_b<<<(4 * 1 * 32 + 255) / 256, blk, 0, stream>>>(We1, We1, 64, 64, 64,  16, 1, 4, fWe1);
  pack_b<<<(4 * 2 * 32 + 255) / 256, blk, 0, stream>>>(We2, We2, 64, 64, 64,  64, 2, 4, fWe2);
  pack_b<<<(4 * 2 * 32 + 255) / 256, blk, 0, stream>>>(Wo1, Wo1, 64, 64, 64,  64, 2, 4, fWo1);

  deg_count<<<(E + 255) / 256, blk, 0, stream>>>(tgt, cnt, E);

  int nTilesN = (N + 15) / 16;
  // Layer 0: y0 = x@Wl0 -> bufY ; z0 = x@Wr0 -> bufZ
  node_gemm<4><<<(nTilesN + 7) / 8, blk, 0, stream>>>(x, 128, fW0, bufY, bufZ, N);
  scatter_add64<<<(unsigned)(((long)E * H + 255) / 256), blk, 0, stream>>>(bufY, src, tgt, bufS, E);
  // h0 = relu(S/cnt + bl0 + z0) -> bufY
  sage_finish<<<(unsigned)(((long)N * H + 255) / 256), blk, 0, stream>>>(bufS, cnt, bl0, bufZ, bufY, N);

  // Layer 1
  zero_f<<<(unsigned)(((long)N * H + 255) / 256), blk, 0, stream>>>(bufS, (long)N * H);
  node_gemm<2><<<(nTilesN + 7) / 8, blk, 0, stream>>>(bufY, 64, fW1, bufZ, bufW, N);
  scatter_add64<<<(unsigned)(((long)E * H + 255) / 256), blk, 0, stream>>>(bufZ, src, tgt, bufS, E);
  // h1 = relu(S/cnt + bl1 + z1) -> bufY
  sage_finish<<<(unsigned)(((long)N * H + 255) / 256), blk, 0, stream>>>(bufS, cnt, bl1, bufW, bufY, N);

  // Edge head
  int nTilesE = (E + 15) / 16;
  edge_head<<<(nTilesE + 7) / 8, blk, 0, stream>>>(ea, src, tgt, bufY, fWe1, be1, fWe2, be2,
                                                   fWo1, bo1, Wo2, bo2, out, E);
}
